// RelationKnowledgeExtractor_47674136986059
// MI455X (gfx1250) — matrix operand order repaired and checked by reading, not compiled, not run on hardware
//
#include <hip/hip_runtime.h>

// Problem constants from the reference: B=512, TD=2048, SD=768, H=128.
#define BSZ 512
#define HSZ 128

typedef __bf16 bf16_t;
typedef __attribute__((ext_vector_type(16))) bf16_t v16bf;
typedef __attribute__((ext_vector_type(8)))  float  v8f;
typedef __attribute__((ext_vector_type(4)))  unsigned int v4u;
typedef __attribute__((ext_vector_type(8)))  int v8i;
typedef __attribute__((ext_vector_type(4)))  int v4i;

union BF16Frag {
  v16bf v;
  unsigned short s[16];
};

#if defined(__has_builtin)
#  if __has_builtin(__builtin_amdgcn_tensor_load_to_lds)
#    define HAVE_TDM 1
#  endif
#endif
#ifndef HAVE_TDM
#  define HAVE_TDM 0
#endif

__device__ __forceinline__ unsigned short f32_to_bf16_bits(float f) {
  union { float f; unsigned u; } c;
  c.f = f;
  unsigned r = c.u + 0x7FFFu + ((c.u >> 16) & 1u);  // round-to-nearest-even
  return (unsigned short)(r >> 16);
}

#if HAVE_TDM
// Build a D# (ISA cdna5 sec 8.3/8.4) for a 2D bf16 tile load Global->LDS with
// LDS destination padding (pad_amount dwords every pad_interval bytes) to
// stagger LDS banks for the WMMA fragment reads.
__device__ __forceinline__ void tdm_load_2d_bf16(
    unsigned lds_byte_off, const unsigned short* gptr,
    int tensor_w, int tensor_h, int row_stride_elems,
    int tile_w, int tile_h, int pad_interval_code, int pad_amount_code) {
  unsigned long long ga = (unsigned long long)gptr;
  v4u g0;
  g0[0] = 1u;                                   // count=1, user-mode descriptor
  g0[1] = lds_byte_off;                         // lds_addr [63:32]
  g0[2] = (unsigned)(ga & 0xFFFFFFFFu);         // global_addr [95:64]
  g0[3] = (unsigned)((ga >> 32) & 0x01FFFFFFu)  // global_addr [120:96]
        | (2u << 30);                           // type=2 ("image")
  v8i g1;
  g1[0] = (int)((1u << 16)                      // data_size = 1 (2 bytes)
              | (1u << 20)                      // pad_enable
              | ((unsigned)pad_interval_code << 22)
              | ((unsigned)pad_amount_code << 25));
  g1[1] = (int)(((unsigned)tensor_w & 0xFFFFu) << 16);           // tensor_dim0[15:0]
  g1[2] = (int)((((unsigned)tensor_w >> 16) & 0xFFFFu)
              | (((unsigned)tensor_h & 0xFFFFu) << 16));         // dim0 hi | dim1 lo
  g1[3] = (int)((((unsigned)tensor_h >> 16) & 0xFFFFu)
              | ((unsigned)tile_w << 16));                       // dim1 hi | tile_dim0
  g1[4] = (int)(unsigned)tile_h;                                 // tile_dim1
  g1[5] = (int)(unsigned)row_stride_elems;                       // tensor_dim0_stride lo
  g1[6] = 0;
  g1[7] = 0;
  v4i g2 = {0, 0, 0, 0};                        // group2: unused (2D tensor)
  v4i g3 = {0, 0, 0, 0};                        // group3: unused (2D tensor)
  v8i g4 = {0, 0, 0, 0, 0, 0, 0, 0};            // extra group (clang-23 6-arg form)
  __builtin_amdgcn_tensor_load_to_lds(g0, g1, g2, g3, g4, 0);
}
#endif

// ---- prep kernels: one-time fp32 -> bf16 conversion -------------------------

__global__ __launch_bounds__(256) void convert_bf16(
    const float* __restrict__ src, unsigned short* __restrict__ dst) {
  const int i = (blockIdx.x * 256 + threadIdx.x) * 4;
  const float4 v = *(const float4*)(src + i);
  ushort4 o;
  o.x = f32_to_bf16_bits(v.x);
  o.y = f32_to_bf16_bits(v.y);
  o.z = f32_to_bf16_bits(v.z);
  o.w = f32_to_bf16_bits(v.w);
  *(ushort4*)(dst + i) = o;
}

// W [K,128] row-major fp32  ->  WT [128][K] bf16 (so B fragments are contiguous)
__global__ __launch_bounds__(256) void transpose_convert(
    const float* __restrict__ W, unsigned short* __restrict__ WT, int K) {
  const int idx = blockIdx.x * 256 + threadIdx.x;  // over K*128
  const int k = idx >> 7, n = idx & 127;
  WT[n * K + k] = f32_to_bf16_bits(W[idx]);
}

// ---- GEMM: C[512,128] = A_bf16[512,K] @ W_bf16[K,128] (+bias) ---------------
// One block = one 16-row tile; 8 waves, wave w owns columns [16w,16w+16).
// A chunk (16 x kc bf16) staged into LDS by the Tensor Data Mover (wave 0),
// with TDM LDS-padding of 4 dwords per row to avoid ds_load_b128 bank
// conflicts. Output fp32 (Cf) or bf16 (Cbf, for tensors re-used as A).
__global__ __launch_bounds__(256) void gemm_bf16_wmma(
    const unsigned short* __restrict__ A, const unsigned short* __restrict__ WT,
    const float* __restrict__ bias, float* __restrict__ Cf,
    unsigned short* __restrict__ Cbf, int K) {
  __shared__ unsigned short ldsA[16 * 264];  // 16 rows x (kc + 8) ushorts max

  const int tid  = threadIdx.x;
  const int lane = tid & 31;
  const int wave = tid >> 5;     // 0..7 -> column tile
  const int half = lane >> 4;    // lane group 0/1
  const int l16  = lane & 15;
  const int row0 = blockIdx.x * 16;
  const int n    = wave * 16 + l16;

  const unsigned lds_base = (unsigned)(unsigned long long)&ldsA[0];  // addr[31:0] = LDS offset
  (void)lds_base;

  v8f acc = {0.f, 0.f, 0.f, 0.f, 0.f, 0.f, 0.f, 0.f};

  for (int k0 = 0; k0 < K; k0 += 256) {
    const int kc = (K - k0 < 256) ? (K - k0) : 256;  // 256 or 128 here
    const int ldstride = kc + 8;                     // ushorts per padded LDS row
    __syncthreads();  // previous chunk fully consumed
#if HAVE_TDM
    if (wave == 0) {
      // pad every row (kc*2 bytes): interval code: 512B->6, 256B->5; amount 4 dwords -> 3
      const int icode = (kc == 256) ? 6 : 5;
      tdm_load_2d_bf16(lds_base, A + (size_t)row0 * K + k0,
                       /*tensor_w=*/K, /*tensor_h=*/BSZ, /*row_stride=*/K,
                       /*tile_w=*/kc, /*tile_h=*/16, icode, 3);
      __builtin_amdgcn_s_wait_tensorcnt(0);
    }
#else
    // fallback: cooperative b128 copy into padded LDS
    for (int idx = tid; idx < 16 * (kc >> 3); idx += 256) {
      const int r = idx / (kc >> 3), c8 = (idx - r * (kc >> 3)) << 3;
      *(uint4*)&ldsA[r * ldstride + c8] =
          *(const uint4*)&A[(size_t)(row0 + r) * K + k0 + c8];
    }
#endif
    if (k0 + 256 < K)  // speculative prefetch of next WT chunk (global_prefetch_b8)
      __builtin_prefetch(&WT[(size_t)n * K + k0 + 256 + half * 128], 0, 0);
    __syncthreads();

    for (int ks = 0; ks < kc; ks += 32) {
      BF16Frag a, b;
      // A fragment per ISA 16-bit 16x32 layout: lane half 0 -> K {0..7,16..23},
      // half 1 -> K {8..15,24..31}; two contiguous 8-elem runs = 2x ds_load_b128
      const int ka = ks + half * 8;
      *(uint4*)&a.s[0] = *(const uint4*)&ldsA[l16 * ldstride + ka];
      *(uint4*)&a.s[8] = *(const uint4*)&ldsA[l16 * ldstride + ka + 16];
      // B fragment: lane = column n, same K order as A: {0..7,16..23} / {8..15,24..31}
      const int kb = k0 + ks + half * 8;
      *(uint4*)&b.s[0] = *(const uint4*)&WT[(size_t)n * K + kb];
      *(uint4*)&b.s[8] = *(const uint4*)&WT[(size_t)n * K + kb + 16];

      acc = __builtin_amdgcn_wmma_f32_16x16x32_bf16(
          /*neg_a=*/false, a.v, /*neg_b=*/false, b.v,
          /*c_mod=*/(short)0, acc, /*reuse_a=*/false, /*reuse_b=*/false);
    }
  }

  // C/D layout: VGPR r -> M = r + 8*half, N = lane&15 (within tile)
  const float bv = bias ? bias[n] : 0.0f;
  if (Cbf) {
#pragma unroll
    for (int r = 0; r < 8; ++r)
      Cbf[(row0 + r + half * 8) * HSZ + n] = f32_to_bf16_bits(acc[r] + bv);
  } else {
#pragma unroll
    for (int r = 0; r < 8; ++r)
      Cf[(row0 + r + half * 8) * HSZ + n] = acc[r] + bv;
  }
}

// ---- pairwise stage ---------------------------------------------------------
// block (bi,bj) -> 16x16 (i,j) tile, one pair per thread:
// rel = sigmoid(dot(relu(U[i]+V[j]), w2) + b2) for both nets; (rs-rt)^2.
__global__ __launch_bounds__(256) void pairwise_loss(
    const float* __restrict__ Ut, const float* __restrict__ Vt,
    const float* __restrict__ Us, const float* __restrict__ Vs,
    const float* __restrict__ W2, const float* __restrict__ b2,
    float* __restrict__ partials) {
  // stride 132 floats: float4-aligned (132%4==0) and bank-staggered
  __shared__ float shUt[16][132];
  __shared__ float shVt[16][132];
  __shared__ float shUs[16][132];
  __shared__ float shVs[16][132];
  __shared__ float shw2[128];
  __shared__ float wred[8];

  const int tid = threadIdx.x;
  const int i0 = blockIdx.x * 16;
  const int j0 = blockIdx.y * 16;

  for (int idx = tid; idx < 16 * 128; idx += 256) {
    const int r = idx >> 7, c = idx & 127;
    shUt[r][c] = Ut[(i0 + r) * HSZ + c];
    shVt[r][c] = Vt[(j0 + r) * HSZ + c];
    shUs[r][c] = Us[(i0 + r) * HSZ + c];
    shVs[r][c] = Vs[(j0 + r) * HSZ + c];
  }
  if (tid < 128) shw2[tid] = W2[tid];
  __syncthreads();

  const int ti = tid >> 4;
  const int tj = tid & 15;
  const float4* ut = (const float4*)&shUt[ti][0];
  const float4* vt = (const float4*)&shVt[tj][0];
  const float4* us = (const float4*)&shUs[ti][0];
  const float4* vs = (const float4*)&shVs[tj][0];
  const float4* w2 = (const float4*)&shw2[0];

  float accT = 0.f, accS = 0.f;
#pragma unroll 8
  for (int h = 0; h < 32; ++h) {
    const float4 u = ut[h], v = vt[h], a = us[h], b = vs[h], w = w2[h];
    float x;
    x = fmaxf(u.x + v.x, 0.f); accT = fmaf(x, w.x, accT);
    x = fmaxf(u.y + v.y, 0.f); accT = fmaf(x, w.y, accT);
    x = fmaxf(u.z + v.z, 0.f); accT = fmaf(x, w.z, accT);
    x = fmaxf(u.w + v.w, 0.f); accT = fmaf(x, w.w, accT);
    x = fmaxf(a.x + b.x, 0.f); accS = fmaf(x, w.x, accS);
    x = fmaxf(a.y + b.y, 0.f); accS = fmaf(x, w.y, accS);
    x = fmaxf(a.z + b.z, 0.f); accS = fmaf(x, w.z, accS);
    x = fmaxf(a.w + b.w, 0.f); accS = fmaf(x, w.w, accS);
  }
  const float b2c = b2[0];
  const float rt = 1.f / (1.f + __expf(-(accT + b2c)));
  const float rs = 1.f / (1.f + __expf(-(accS + b2c)));
  const float d = rs - rt;
  float val = ((i0 + ti) == (j0 + tj)) ? 0.f : d * d;

#pragma unroll
  for (int off = 16; off > 0; off >>= 1) val += __shfl_down(val, off, 32);
  if ((tid & 31) == 0) wred[tid >> 5] = val;
  __syncthreads();
  if (tid == 0) {
    float s = 0.f;
#pragma unroll
    for (int w = 0; w < 8; ++w) s += wred[w];
    partials[blockIdx.y * 32 + blockIdx.x] = s;
  }
}

__global__ __launch_bounds__(256) void final_reduce(
    const float* __restrict__ partials, float* __restrict__ out) {
  __shared__ float sh[256];
  const int tid = threadIdx.x;
  float s = 0.f;
  for (int i = tid; i < 1024; i += 256) s += partials[i];
  sh[tid] = s;
  __syncthreads();
  for (int off = 128; off > 0; off >>= 1) {
    if (tid < off) sh[tid] += sh[tid + off];
    __syncthreads();
  }
  if (tid == 0) out[0] = sh[0] * (1.0f / ((float)BSZ * (float)BSZ));
}

extern "C" void kernel_launch(void* const* d_in, const int* in_sizes, int n_in,
                              void* d_out, int out_size, void* d_ws, size_t ws_size,
                              hipStream_t stream) {
  const float* Tf = (const float*)d_in[0];   // [512, 2048]
  const float* Sf = (const float*)d_in[1];   // [512, 768]
  const float* Wt = (const float*)d_in[2];   // [2048, 128]
  const float* bt = (const float*)d_in[3];   // [128]
  const float* Ws = (const float*)d_in[4];   // [768, 128]
  const float* bs = (const float*)d_in[5];   // [128]
  const float* W1 = (const float*)d_in[6];   // [256, 128]
  const float* b1 = (const float*)d_in[7];   // [128]
  const float* W2 = (const float*)d_in[8];   // [128, 1]
  const float* b2 = (const float*)d_in[9];   // [1]
  float* out = (float*)d_out;

  // workspace layout (bf16 region first, then fp32; boundary is 16B aligned)
  unsigned short* Tf_bf = (unsigned short*)d_ws;       // 512*2048
  unsigned short* Sf_bf = Tf_bf + 512 * 2048;          // 512*768
  unsigned short* WtT   = Sf_bf + 512 * 768;           // 128*2048
  unsigned short* WsT   = WtT + 128 * 2048;            // 128*768
  unsigned short* W1Tt  = WsT + 128 * 768;             // 128*128
  unsigned short* W1Tb  = W1Tt + 128 * 128;            // 128*128
  unsigned short* tp_bf = W1Tb + 128 * 128;            // 512*128
  unsigned short* sp_bf = tp_bf + 512 * 128;           // 512*128
  float* Ut = (float*)(sp_bf + 512 * 128);             // 512*128 fp32
  float* Vt = Ut + BSZ * HSZ;
  float* Us = Vt + BSZ * HSZ;
  float* Vs = Us + BSZ * HSZ;
  float* partials = Vs + BSZ * HSZ;                    // 1024

  // one-time precision conversion / weight transpose
  convert_bf16<<<(512 * 2048) / 1024, 256, 0, stream>>>(Tf, Tf_bf);
  convert_bf16<<<(512 * 768) / 1024, 256, 0, stream>>>(Sf, Sf_bf);
  transpose_convert<<<(2048 * 128) / 256, 256, 0, stream>>>(Wt, WtT, 2048);
  transpose_convert<<<(768 * 128) / 256, 256, 0, stream>>>(Ws, WsT, 768);
  transpose_convert<<<(128 * 128) / 256, 256, 0, stream>>>(W1, W1Tt, 128);
  transpose_convert<<<(128 * 128) / 256, 256, 0, stream>>>(W1 + 128 * HSZ, W1Tb, 128);

  // projections (bf16 out: consumed only as A operands)
  gemm_bf16_wmma<<<32, 256, 0, stream>>>(Tf_bf, WtT, bt, nullptr, tp_bf, 2048);
  gemm_bf16_wmma<<<32, 256, 0, stream>>>(Sf_bf, WsT, bs, nullptr, sp_bf, 768);
  // rank factors of the pairwise MLP first layer (bias folded into V)
  gemm_bf16_wmma<<<32, 256, 0, stream>>>(tp_bf, W1Tt, nullptr, Ut, nullptr, 128);
  gemm_bf16_wmma<<<32, 256, 0, stream>>>(tp_bf, W1Tb, b1,      Vt, nullptr, 128);
  gemm_bf16_wmma<<<32, 256, 0, stream>>>(sp_bf, W1Tt, nullptr, Us, nullptr, 128);
  gemm_bf16_wmma<<<32, 256, 0, stream>>>(sp_bf, W1Tb, b1,      Vs, nullptr, 128);
  // pairwise nonlinearity + MSE (deterministic two-pass reduction)
  pairwise_loss<<<dim3(32, 32), 256, 0, stream>>>(Ut, Vt, Us, Vs, W2, b2, partials);
  final_reduce<<<1, 256, 0, stream>>>(partials, out);
}
